// DecoderStack_26929444946350
// MI455X (gfx1250) — matrix operand order repaired and checked by reading, not compiled, not run on hardware
//
#include <hip/hip_runtime.h>
#include <hip/hip_bf16.h>
#include <cstdint>
#include <cstddef>

// ---------------------------------------------------------------------------
// Types for CDNA5 WMMA (wave32): v16bf A/B fragments, v8f accumulator.
// ---------------------------------------------------------------------------
typedef __bf16 bf16x8  __attribute__((ext_vector_type(8)));
typedef __bf16 bf16x16 __attribute__((ext_vector_type(16)));
typedef float  f32x8   __attribute__((ext_vector_type(8)));

__device__ __forceinline__ unsigned short f2bf(float f) {
  union { float f; unsigned u; } v; v.f = f;
  unsigned r = v.u + 0x7FFFu + ((v.u >> 16) & 1u);   // round-to-nearest-even
  return (unsigned short)(r >> 16);
}

// ---------------------------------------------------------------------------
// Tensor Data Mover staging (gfx1250). D# packing per CDNA5 ISA.
// This toolchain exposes the 6-arg builtin:
//   (uint32x4 g0, int32x8 g1, int32x4 g2, int32x4 g3, int32x8 extra, i32 cpol)
// Guarded by __has_builtin so the host pass falls back to the manual staging
// path. Template arguments never depend on this macro, only function bodies,
// so host/device instantiations stay symbol-identical.
// ---------------------------------------------------------------------------
#if defined(__HIP_DEVICE_COMPILE__) && __has_builtin(__builtin_amdgcn_tensor_load_to_lds)
#define HAS_TDM 1
#else
#define HAS_TDM 0
#endif

typedef unsigned int u32x4 __attribute__((ext_vector_type(4)));
typedef int          i32x8 __attribute__((ext_vector_type(8)));
typedef int          i32x4 __attribute__((ext_vector_type(4)));

// Copy a 128-row x 32-col bf16 tile (row stride = ld elements) from global to
// LDS at byte offset lds_off, padded to 40-element rows (16 DW data + 4 DW pad).
__device__ __forceinline__ void tdm_tile_bf16(unsigned lds_off, const void* gtile,
                                              unsigned tensor_d0, unsigned tensor_d1,
                                              unsigned ld)
{
#if HAS_TDM
  const unsigned long long ga = (unsigned long long)gtile;
  u32x4 g0;
  g0[0] = 1u;                                            // count=1, user D#
  g0[1] = lds_off;                                       // lds_addr (bytes)
  g0[2] = (unsigned)(ga & 0xFFFFFFFFull);                // global_addr[31:0]
  g0[3] = (unsigned)((ga >> 32) & 0x01FFFFFFull)         // global_addr[56:32]
        | 0x80000000u;                                   // type = 2 ("image")
  i32x8 g1;
  g1[0] = (1 << 16)        // data_size = 1 -> 2 bytes
        | (1 << 20)        // pad_enable
        | (3 << 22)        // pad_interval: 16 DWORDs (32 bf16) of data per row
        | (3 << 25);       // pad_amount:    4 DWORDs (8 bf16) pad -> 40-elem pitch
  g1[1] = (int)((tensor_d0 & 0xFFFFu) << 16);            // tensor_dim0[15:0]
  g1[2] = (int)((tensor_d0 >> 16) | ((tensor_d1 & 0xFFFFu) << 16)); // dim0 hi | dim1 lo
  g1[3] = (int)((tensor_d1 >> 16) | (32u << 16));        // dim1 hi | tile_dim0 = 32
  g1[4] = 128;                                           // tile_dim1 = 128, tile_dim2 = 0
  g1[5] = (int)ld;                                       // tensor_dim0_stride[31:0]
  g1[6] = 0;
  g1[7] = 0;
  const i32x4 z4 = {0, 0, 0, 0};                         // groups 2/3 unused (2-D tensor)
  const i32x8 z8 = {0, 0, 0, 0, 0, 0, 0, 0};
  __builtin_amdgcn_tensor_load_to_lds(g0, g1, z4, z4, z8, 0);
#else
  (void)lds_off; (void)gtile; (void)tensor_d0; (void)tensor_d1; (void)ld;
#endif
}

__device__ __forceinline__ void tdm_wait0()
{
#if HAS_TDM
  __builtin_amdgcn_s_wait_tensorcnt(0);
#endif
}

// ---------------------------------------------------------------------------
// Tiled WMMA GEMM:  C[M,N] = epilogue( A[M,K] * B * scale + bias )
//   TA/TB : float (converted to bf16 on LDS fill) or unsigned short (bf16 bits)
//   BT    : false -> B stored row-major [K,N]; true -> B stored [N,K] (B^T)
//   EPI   : 0 = scale only, 1 = scale + bias, 2 = scale + bias + exact GELU
//   TOUT  : float or unsigned short (bf16)
//   TDMA/TDMB: stage that operand with the Tensor Data Mover (bf16 src, and
//              for B only in BT layout — TDM cannot transpose or convert).
// Block: 128 threads = 4 waves (2x2), each wave computes 64x64 via 16 WMMAs.
// ---------------------------------------------------------------------------
constexpr int BM = 128, BN = 128, BK = 32, PIT = 40;   // PIT*2B = 80B rows, 16B aligned

template <typename TA, typename TB, bool BT, int EPI, typename TOUT,
          bool TDMA = false, bool TDMB = false>
__global__ __launch_bounds__(128)
void gemm_wmma(const TA* __restrict__ A, const TB* __restrict__ B,
               const float* __restrict__ bias, TOUT* __restrict__ C,
               int M, int N, int K, int lda, int ldb, int ldc, float scale)
{
  __shared__ unsigned short As[BM * PIT];   // [m][k], bf16 bits
  __shared__ unsigned short Bs[BN * PIT];   // [n][k], bf16 bits (B transposed in LDS)

  // Per compile pass: use TDM only where available; bodies differ, symbols don't.
  constexpr bool useTdmA = TDMA && (HAS_TDM != 0) && (sizeof(TA) == 2);
  constexpr bool useTdmB = TDMB && (HAS_TDM != 0) && (sizeof(TB) == 2) && BT;

  const int t    = threadIdx.x;
  const int m0   = blockIdx.y * BM;
  const int n0   = blockIdx.x * BN;
  const int w    = t >> 5;
  const int lane = t & 31;
  const int wm   = (w >> 1) * 64;
  const int wn   = (w & 1) * 64;
  const int l16  = lane & 15;
  const int lh   = lane >> 4;               // K-half select per ISA 16-bit layout

  const unsigned as_off = (unsigned)(uintptr_t)(&As[0]);   // LDS byte offsets
  const unsigned bs_off = (unsigned)(uintptr_t)(&Bs[0]);

  f32x8 acc[16] = {};

  for (int k0 = 0; k0 < K; k0 += BK) {
    // ---- TDM DMA staging (wave 0 issues; EXEC-independent engine copy) ----
    if constexpr (useTdmA || useTdmB) {
      if (t < 32) {
        if constexpr (useTdmA)
          tdm_tile_bf16(as_off, A + (size_t)m0 * lda + k0, (unsigned)K, (unsigned)M,
                        (unsigned)lda);
        if constexpr (useTdmB)
          tdm_tile_bf16(bs_off, B + (size_t)n0 * ldb + k0, (unsigned)K, (unsigned)N,
                        (unsigned)ldb);
      }
    }
    // ---- manual staging: A tile (BM x BK), coalesced, fp32->bf16 if needed ----
    if constexpr (!useTdmA) {
#pragma unroll
      for (int j = 0; j < 8; ++j) {
        const int e = (t + j * 128) * 4;
        const int m = e >> 5, k = e & 31;
        if (j == 0 && k0 + BK < K)
          __builtin_prefetch(A + (size_t)(m0 + m) * lda + (k0 + BK + k), 0, 1);
        unsigned short b0, b1, b2, b3;
        if constexpr (sizeof(TA) == 4) {
          const float4 f = *reinterpret_cast<const float4*>(A + (size_t)(m0 + m) * lda + (k0 + k));
          b0 = f2bf(f.x); b1 = f2bf(f.y); b2 = f2bf(f.z); b3 = f2bf(f.w);
        } else {
          const ushort4 u = *reinterpret_cast<const ushort4*>(A + (size_t)(m0 + m) * lda + (k0 + k));
          b0 = u.x; b1 = u.y; b2 = u.z; b3 = u.w;
        }
        *reinterpret_cast<ushort4*>(&As[m * PIT + k]) = make_ushort4(b0, b1, b2, b3);
      }
    }
    // ---- manual staging: B tile into [n][k] layout ----
    if constexpr (BT) {
      if constexpr (!useTdmB) {
        // B is [N,K]: contiguous along k, direct copy
#pragma unroll
        for (int j = 0; j < 8; ++j) {
          const int e = (t + j * 128) * 4;
          const int n = e >> 5, k = e & 31;
          unsigned short b0, b1, b2, b3;
          if constexpr (sizeof(TB) == 4) {
            const float4 f = *reinterpret_cast<const float4*>(B + (size_t)(n0 + n) * ldb + (k0 + k));
            b0 = f2bf(f.x); b1 = f2bf(f.y); b2 = f2bf(f.z); b3 = f2bf(f.w);
          } else {
            const ushort4 u = *reinterpret_cast<const ushort4*>(B + (size_t)(n0 + n) * ldb + (k0 + k));
            b0 = u.x; b1 = u.y; b2 = u.z; b3 = u.w;
          }
          *reinterpret_cast<ushort4*>(&Bs[n * PIT + k]) = make_ushort4(b0, b1, b2, b3);
        }
      }
    } else {
      // B is [K,N]: coalesced along n, transposed scatter into LDS (TDM can't transpose)
#pragma unroll
      for (int j = 0; j < 8; ++j) {
        const int e = (t + j * 128) * 4;
        const int k = e >> 7, n = e & 127;
        if (j == 0 && k0 + BK < K)
          __builtin_prefetch(B + (size_t)(k0 + BK + k) * ldb + (n0 + n), 0, 1);
        unsigned short b0, b1, b2, b3;
        if constexpr (sizeof(TB) == 4) {
          const float4 f = *reinterpret_cast<const float4*>(B + (size_t)(k0 + k) * ldb + (n0 + n));
          b0 = f2bf(f.x); b1 = f2bf(f.y); b2 = f2bf(f.z); b3 = f2bf(f.w);
        } else {
          const ushort4 u = *reinterpret_cast<const ushort4*>(B + (size_t)(k0 + k) * ldb + (n0 + n));
          b0 = u.x; b1 = u.y; b2 = u.z; b3 = u.w;
        }
        Bs[(n + 0) * PIT + k] = b0;
        Bs[(n + 1) * PIT + k] = b1;
        Bs[(n + 2) * PIT + k] = b2;
        Bs[(n + 3) * PIT + k] = b3;
      }
    }
    // Drain TDM before making the tile visible to all waves.
    if constexpr (useTdmA || useTdmB) {
      if (t < 32) tdm_wait0();
    }
    __syncthreads();

    // ---- fragments per the CDNA5 16-bit WMMA VGPR layouts ----
    // A 16x32: lanes<16 hold K 0..7 & 16..23, lanes>=16 hold K 8..15 & 24..31
    bf16x16 af[4], bfr[4];
#pragma unroll
    for (int mt = 0; mt < 4; ++mt) {
      const unsigned short* pr = &As[(wm + mt * 16 + l16) * PIT];
      const bf16x8 lo = *reinterpret_cast<const bf16x8*>(pr + lh * 8);
      const bf16x8 hi = *reinterpret_cast<const bf16x8*>(pr + 16 + lh * 8);
      af[mt] = __builtin_shufflevector(lo, hi, 0,1,2,3,4,5,6,7,8,9,10,11,12,13,14,15);
    }
    // B 32x16: lanes<16 hold K 0..7 & 16..23, lanes>=16 hold K 8..15 & 24..31 (column = lane%16)
#pragma unroll
    for (int nt = 0; nt < 4; ++nt) {
      const unsigned short* pr = &Bs[(wn + nt * 16 + l16) * PIT];
      const bf16x8 lo = *reinterpret_cast<const bf16x8*>(pr + lh * 8);
      const bf16x8 hi = *reinterpret_cast<const bf16x8*>(pr + 16 + lh * 8);
      bfr[nt] = __builtin_shufflevector(lo, hi, 0,1,2,3,4,5,6,7,8,9,10,11,12,13,14,15);
    }
#pragma unroll
    for (int mt = 0; mt < 4; ++mt)
#pragma unroll
      for (int nt = 0; nt < 4; ++nt)
        acc[mt * 4 + nt] = __builtin_amdgcn_wmma_f32_16x16x32_bf16(
            false, af[mt], false, bfr[nt], (short)0, acc[mt * 4 + nt], false, false);
    __syncthreads();
  }

  // ---- epilogue + store (C layout: VGPR r -> row r + 8*lh, col = lane%16) ----
#pragma unroll
  for (int mt = 0; mt < 4; ++mt) {
#pragma unroll
    for (int nt = 0; nt < 4; ++nt) {
      const int gn = n0 + wn + nt * 16 + l16;
      float bterm = 0.0f;
      if constexpr (EPI >= 1) bterm = bias[gn];
      const f32x8 a = acc[mt * 4 + nt];
#pragma unroll
      for (int r = 0; r < 8; ++r) {
        const int gm = m0 + wm + mt * 16 + lh * 8 + r;
        float v = a[r] * scale + bterm;
        if constexpr (EPI == 2) v = 0.5f * v * (1.0f + erff(v * 0.70710678118f));
        if constexpr (sizeof(TOUT) == 4) C[(size_t)gm * ldc + gn] = v;
        else                             C[(size_t)gm * ldc + gn] = f2bf(v);
      }
    }
  }
}

// ---------------------------------------------------------------------------
// Row softmax over T columns, bf16 probability output. One block (256 thr) / row.
// ---------------------------------------------------------------------------
__global__ __launch_bounds__(256)
void softmax_bf16(const float* __restrict__ S, unsigned short* __restrict__ P, int T)
{
  const int row = blockIdx.x, t = threadIdx.x;
  const float* sr = S + (size_t)row * T;
  float v[8];
  float mx = -3.4e38f;
#pragma unroll
  for (int j = 0; j < 8; ++j) { v[j] = sr[t + j * 256]; mx = fmaxf(mx, v[j]); }
#pragma unroll
  for (int o = 16; o > 0; o >>= 1) mx = fmaxf(mx, __shfl_xor(mx, o, 32));
  __shared__ float sm[8];
  const int lane = t & 31, wv = t >> 5;
  if (lane == 0) sm[wv] = mx;
  __syncthreads();
  float m = sm[0];
#pragma unroll
  for (int i = 1; i < 8; ++i) m = fmaxf(m, sm[i]);
  float s = 0.f;
#pragma unroll
  for (int j = 0; j < 8; ++j) { v[j] = __expf(v[j] - m); s += v[j]; }
#pragma unroll
  for (int o = 16; o > 0; o >>= 1) s += __shfl_xor(s, o, 32);
  __shared__ float ss[8];
  if (lane == 0) ss[wv] = s;
  __syncthreads();
  float tot = 0.f;
#pragma unroll
  for (int i = 0; i < 8; ++i) tot += ss[i];
  const float inv = 1.0f / tot;
  unsigned short* pr = P + (size_t)row * T;
#pragma unroll
  for (int j = 0; j < 8; ++j) pr[t + j * 256] = f2bf(v[j] * inv);
}

// ---------------------------------------------------------------------------
// Block reductions + layernorm kernels (D = 1024, 256 threads, 4 elems/thread)
// ---------------------------------------------------------------------------
__device__ __forceinline__ void block_sum2(float& a, float& b, float* sh)
{
#pragma unroll
  for (int o = 16; o > 0; o >>= 1) { a += __shfl_xor(a, o, 32); b += __shfl_xor(b, o, 32); }
  const int lane = threadIdx.x & 31, wv = threadIdx.x >> 5;
  if (lane == 0) { sh[2 * wv] = a; sh[2 * wv + 1] = b; }
  __syncthreads();
  float sa = 0.f, sb = 0.f;
#pragma unroll
  for (int i = 0; i < 8; ++i) { sa += sh[2 * i]; sb += sh[2 * i + 1]; }
  __syncthreads();
  a = sa; b = sb;
}

// out = LN( LN(attn + xin, g1,b1) + xin, g2,b2 )  -- fused inner+outer LN
__global__ __launch_bounds__(256)
void double_ln(const float* __restrict__ attn, const float* __restrict__ xin,
               const float* __restrict__ g1, const float* __restrict__ b1,
               const float* __restrict__ g2, const float* __restrict__ b2,
               float* __restrict__ out)
{
  const int row = blockIdx.x, t = threadIdx.x;
  const float* ar = attn + (size_t)row * 1024;
  const float* xr = xin + (size_t)row * 1024;
  __shared__ float sh1[16], sh2[16];
  float u[4], xv[4], wv[4];
  float s = 0.f, q = 0.f;
#pragma unroll
  for (int j = 0; j < 4; ++j) {
    const int c = t + j * 256;
    xv[j] = xr[c];
    u[j] = ar[c] + xv[j];
    s += u[j]; q += u[j] * u[j];
  }
  block_sum2(s, q, sh1);
  float mean = s * (1.0f / 1024.0f);
  float var  = q * (1.0f / 1024.0f) - mean * mean;
  float rs   = rsqrtf(var + 1e-5f);
  s = 0.f; q = 0.f;
#pragma unroll
  for (int j = 0; j < 4; ++j) {
    const int c = t + j * 256;
    const float tt = (u[j] - mean) * rs * g1[c] + b1[c];
    wv[j] = tt + xv[j];
    s += wv[j]; q += wv[j] * wv[j];
  }
  block_sum2(s, q, sh2);
  mean = s * (1.0f / 1024.0f);
  var  = q * (1.0f / 1024.0f) - mean * mean;
  rs   = rsqrtf(var + 1e-5f);
#pragma unroll
  for (int j = 0; j < 4; ++j) {
    const int c = t + j * 256;
    out[(size_t)row * 1024 + c] = (wv[j] - mean) * rs * g2[c] + b2[c];
  }
}

// out = LN(h + res, g, b)
__global__ __launch_bounds__(256)
void ln_residual(const float* __restrict__ h, const float* __restrict__ res,
                 const float* __restrict__ g, const float* __restrict__ b,
                 float* __restrict__ out)
{
  const int row = blockIdx.x, t = threadIdx.x;
  const float* hr = h + (size_t)row * 1024;
  const float* rr = res + (size_t)row * 1024;
  __shared__ float sh[16];
  float u[4];
  float s = 0.f, q = 0.f;
#pragma unroll
  for (int j = 0; j < 4; ++j) {
    const int c = t + j * 256;
    u[j] = hr[c] + rr[c];
    s += u[j]; q += u[j] * u[j];
  }
  block_sum2(s, q, sh);
  const float mean = s * (1.0f / 1024.0f);
  const float var  = q * (1.0f / 1024.0f) - mean * mean;
  const float rs   = rsqrtf(var + 1e-5f);
#pragma unroll
  for (int j = 0; j < 4; ++j) {
    const int c = t + j * 256;
    out[(size_t)row * 1024 + c] = (u[j] - mean) * rs * g[c] + b[c];
  }
}

// ---------------------------------------------------------------------------
// Host-side pipeline
// ---------------------------------------------------------------------------
extern "C" void kernel_launch(void* const* d_in, const int* in_sizes, int n_in,
                              void* d_out, int out_size, void* d_ws, size_t ws_size,
                              hipStream_t stream)
{
  const float* x    = (const float*)d_in[0];
  const float* y    = (const float*)d_in[1];
  const float* wq_m = (const float*)d_in[2];
  const float* bq_m = (const float*)d_in[3];
  const float* wk_m = (const float*)d_in[4];
  const float* bk_m = (const float*)d_in[5];
  const float* wv_m = (const float*)d_in[6];
  const float* bv_m = (const float*)d_in[7];
  const float* g_m  = (const float*)d_in[8];
  const float* b_m  = (const float*)d_in[9];
  const float* wq_c = (const float*)d_in[10];
  const float* bq_c = (const float*)d_in[11];
  const float* wk_c = (const float*)d_in[12];
  const float* bk_c = (const float*)d_in[13];
  const float* wv_c = (const float*)d_in[14];
  const float* bv_c = (const float*)d_in[15];
  const float* g_c  = (const float*)d_in[16];
  const float* b_c  = (const float*)d_in[17];
  const float* f0_w = (const float*)d_in[18];
  const float* f0_b = (const float*)d_in[19];
  const float* f1_w = (const float*)d_in[20];
  const float* f1_b = (const float*)d_in[21];
  const float* g_d  = (const float*)d_in[22];
  const float* b_d  = (const float*)d_in[23];

  constexpr int Bsz = 4, S = 2048, T = 2048, D = 1024;
  constexpr int MS = Bsz * S;                 // 8192 rows
  constexpr size_t MB = 1024 * 1024;

  char* wsb = (char*)d_ws;                    // ws layout (136 MB peak):
  unsigned short* Qb    = (unsigned short*)(wsb + 0);         // 16 MB bf16
  unsigned short* Kb    = (unsigned short*)(wsb + 16 * MB);   // 16 MB bf16
  unsigned short* Vb    = (unsigned short*)(wsb + 32 * MB);   // 16 MB bf16
  float*          score = (float*)         (wsb + 48 * MB);   // 16 MB fp32 (per batch)
  unsigned short* probs = (unsigned short*)(wsb + 64 * MB);   //  8 MB bf16 (per batch)
  float*          attn  = (float*)         (wsb + 72 * MB);   // 32 MB fp32
  float*          x1    = (float*)         (wsb + 104 * MB);  // 32 MB fp32
  float*          h2    = (float*)         (wsb + 0);         // reuse QKV region in FFN

  const dim3 blk(128);
  const dim3 gProj(D / 128, MS / 128);        // (8, 64)
  const dim3 gScore(T / 128, S / 128);        // (16, 16)
  const dim3 gAV(D / 128, S / 128);           // (8, 16)

  auto attention = [&](const float* qin, const float* kvin,
                       const float* wq, const float* bq,
                       const float* wk, const float* bk,
                       const float* wvp, const float* bv) {
    gemm_wmma<float, float, false, 1, unsigned short><<<gProj, blk, 0, stream>>>(
        qin, wq, bq, Qb, MS, D, D, D, D, D, 1.0f);
    gemm_wmma<float, float, false, 1, unsigned short><<<gProj, blk, 0, stream>>>(
        kvin, wk, bk, Kb, MS, D, D, D, D, D, 1.0f);
    gemm_wmma<float, float, false, 1, unsigned short><<<gProj, blk, 0, stream>>>(
        kvin, wvp, bv, Vb, MS, D, D, D, D, D, 1.0f);
    for (int b = 0; b < Bsz; ++b) {
      const unsigned short* Qbb = Qb + (size_t)b * S * D;
      const unsigned short* Kbb = Kb + (size_t)b * T * D;
      const unsigned short* Vbb = Vb + (size_t)b * T * D;
      float* attnb = attn + (size_t)b * S * D;
      // scores = (Q @ K^T) * SCALE : both tiles DMA'd by the Tensor Data Mover
      gemm_wmma<unsigned short, unsigned short, true, 0, float, true, true>
          <<<gScore, blk, 0, stream>>>(
          Qbb, Kbb, nullptr, score, S, T, D, D, D, T, 0.125f);
      softmax_bf16<<<S, 256, 0, stream>>>(score, probs, T);
      // attn = probs @ V : probs tile via TDM, V staged manually (transpose)
      gemm_wmma<unsigned short, unsigned short, false, 0, float, true, false>
          <<<gAV, blk, 0, stream>>>(
          probs, Vbb, nullptr, attnb, S, D, T, T, D, D, 1.0f);
    }
  };

  // ---- self attention + fused double LN ----
  attention(x, x, wq_m, bq_m, wk_m, bk_m, wv_m, bv_m);
  double_ln<<<MS, 256, 0, stream>>>(attn, x, g_m, b_m, g_d, b_d, x1);

  // ---- cross attention vs memory y + fused double LN (row-local, in-place safe) ----
  attention(x1, y, wq_c, bq_c, wk_c, bk_c, wv_c, bv_c);
  double_ln<<<MS, 256, 0, stream>>>(attn, x1, g_c, b_c, g_d, b_d, x1);

  // ---- FFN: two GELU GEMMs + residual LN ----
  float* h1 = attn;  // reuse
  gemm_wmma<float, float, false, 2, float><<<gProj, blk, 0, stream>>>(
      x1, f0_w, f0_b, h1, MS, D, D, D, D, D, 1.0f);
  gemm_wmma<float, float, false, 2, float><<<gProj, blk, 0, stream>>>(
      h1, f1_w, f1_b, h2, MS, D, D, D, D, D, 1.0f);
  ln_residual<<<MS, 256, 0, stream>>>(h2, x1, g_d, b_d, (float*)d_out);
}
